// RNN_71382356460196
// MI455X (gfx1250) — matrix operand order repaired and checked by reading, not compiled, not run on hardware
//
#include <hip/hip_runtime.h>
#include <stdint.h>

#define B_ 32
#define T_ 1024
#define I_ 1024
#define H_ 1024
#define NWG 32     // persistent workgroups in recurrent phase (one H-slice each)
#define HS  32     // output channels owned per workgroup

typedef __bf16 bf16_t;
typedef bf16_t v16bf __attribute__((ext_vector_type(16)));
typedef float  v8f   __attribute__((ext_vector_type(8)));

static __device__ __forceinline__ unsigned short f2bf(float f) {
  union { float f; unsigned u; } x; x.f = f;
  unsigned r = x.u + 0x7FFFu + ((x.u >> 16) & 1u);   // round-to-nearest-even
  return (unsigned short)(r >> 16);
}

static __device__ __forceinline__ v8f z8() {
  v8f v; for (int i = 0; i < 8; ++i) v[i] = 0.0f; return v;
}

union FragU { v16bf v; uint4 q[2]; };

// A-matrix 16x32 bf16 fragment from row-major [*, ld] at (row0, k0).
// ISA 16-bit A table: lanes0-15 row=lane hold K 0-7 (v0-3) & 16-23 (v4-7);
// lanes16-31 hold K 8-15 & 24-31. Dword j packs K=2j (lo), 2j+1 (hi).
static __device__ __forceinline__ v16bf load_frag_a(const unsigned short* __restrict__ A,
                                                    int ld, int row0, int k0, int lane) {
  int r  = row0 + (lane & 15);
  int kb = k0 + ((lane >> 4) << 3);
  const unsigned short* p = A + r * ld + kb;
  FragU f;
  f.q[0] = *(const uint4*)(p);        // K kb..kb+7
  f.q[1] = *(const uint4*)(p + 16);   // K kb+16..kb+23
  return f.v;
}

// B-matrix 32x16 bf16 fragment; source W stored [N, K] row-major (= B^T).
// B layout (matches A table): lanes0-15 (col N=lane) hold K 0-7 & 16-23, lanes16-31 hold K 8-15 & 24-31.
static __device__ __forceinline__ v16bf load_frag_b(const unsigned short* __restrict__ W,
                                                    int ld, int n0, int k0, int lane) {
  int r  = n0 + (lane & 15);
  int kb = k0 + ((lane >> 4) << 3);
  const unsigned short* p = W + r * ld + kb;
  FragU f;
  f.q[0] = *(const uint4*)(p);        // K kb..kb+7
  f.q[1] = *(const uint4*)(p + 16);   // K kb+16..kb+23
  return f.v;
}

#define WMMA_BF16(a, b, c) \
  __builtin_amdgcn_wmma_f32_16x16x32_bf16(false, (a), false, (b), (short)0, (c), false, false)

// ---------------------------------------------------------------- conversions
__global__ void cvt_f32_bf16(const float* __restrict__ src,
                             unsigned short* __restrict__ dst, int n) {
  int i = blockIdx.x * blockDim.x + threadIdx.x;
  int stride = gridDim.x * blockDim.x;
  for (; i < n; i += stride) dst[i] = f2bf(src[i]);
}

// --------------------------------------------------- phase 1: input projections
// C[g] = X @ W[g]^T  for g in {r,z,n}.  grid = (256*8, 3), block = 256 (8 waves).
// WG tile 128x128; wave tile 64x32 (4 M-tiles x 2 N-tiles).
__global__ void __launch_bounds__(256, 1)
gemm_inproj(const unsigned short* __restrict__ xbf,
            const unsigned short* __restrict__ w0,
            const unsigned short* __restrict__ w1,
            const unsigned short* __restrict__ w2,
            float* __restrict__ o0, float* __restrict__ o1, float* __restrict__ o2) {
  const unsigned short* W = (blockIdx.y == 0) ? w0 : (blockIdx.y == 1) ? w1 : w2;
  float* O = (blockIdx.y == 0) ? o0 : (blockIdx.y == 1) ? o1 : o2;

  int nblk = blockIdx.x & 7;
  int mblk = blockIdx.x >> 3;
  int wave = threadIdx.x >> 5;
  int lane = threadIdx.x & 31;
  int mbase = mblk * 128 + (wave & 1) * 64;
  int nbase = nblk * 128 + (wave >> 1) * 32;

  v8f acc[4][2];
  for (int i = 0; i < 4; ++i)
    for (int j = 0; j < 2; ++j) acc[i][j] = z8();

  for (int k0 = 0; k0 < I_; k0 += 32) {
    v16bf a[4], b[2];
    for (int i = 0; i < 4; ++i) a[i] = load_frag_a(xbf, I_, mbase + 16 * i, k0, lane);
    for (int j = 0; j < 2; ++j) b[j] = load_frag_b(W, I_, nbase + 16 * j, k0, lane);
    for (int i = 0; i < 4; ++i)
      for (int j = 0; j < 2; ++j) acc[i][j] = WMMA_BF16(a[i], b[j], acc[i][j]);
  }

  int ncol = lane & 15;
  int moff = (lane >> 4) << 3;
  for (int i = 0; i < 4; ++i)
    for (int j = 0; j < 2; ++j)
      for (int rr = 0; rr < 8; ++rr)
        O[(size_t)(mbase + 16 * i + moff + rr) * H_ + (nbase + 16 * j + ncol)] = acc[i][j][rr];
}

// ------------------------------------------------------------- grid barrier
static __device__ __forceinline__ void gbar(unsigned* bar, unsigned target) {
  __threadfence();
  __syncthreads();
  if (threadIdx.x == 0) {
    atomicAdd(bar, 1u);
    while (__hip_atomic_load(bar, __ATOMIC_RELAXED, __HIP_MEMORY_SCOPE_AGENT) < target)
      __builtin_amdgcn_s_sleep(2);
  }
  __syncthreads();
  __threadfence();
}

// ----------------------------------------------- phase 2: persistent recurrence
// 32 WGs x 128 threads (4 waves). WG owns channels [n0, n0+32); its bf16 slices
// of Whr/Whz/Whn live in LDS (192 KB) for all 1024 steps. Per step:
//   phase A: r,z GEMMs over h (stable), write rh = sigmoid(r)*h  -> barrier
//   phase B: n GEMM over rh, h_new = (1-z)h + z*tanh(n)          -> barrier
__global__ void __launch_bounds__(128, 1)
gru_recurrent(const float* __restrict__ xr, const float* __restrict__ xz,
              const float* __restrict__ xn,
              const float* __restrict__ bhr, const float* __restrict__ bhz,
              const float* __restrict__ bhn,
              const unsigned short* __restrict__ whr,
              const unsigned short* __restrict__ whz,
              const unsigned short* __restrict__ whn,
              float* __restrict__ hf, unsigned short* __restrict__ hbf,
              unsigned short* __restrict__ rhbf, unsigned* __restrict__ bar,
              float* __restrict__ out, float* __restrict__ hn) {
  extern __shared__ unsigned short lds[];
  unsigned short* l_whr = lds;
  unsigned short* l_whz = lds + HS * H_;
  unsigned short* l_whn = lds + 2 * HS * H_;
  const int n0 = blockIdx.x * HS;

  { // stage weight slices (rows n0..n0+HS-1 are contiguous in [H,K] row-major)
    const unsigned* s0 = (const unsigned*)(whr + (size_t)n0 * H_);
    const unsigned* s1 = (const unsigned*)(whz + (size_t)n0 * H_);
    const unsigned* s2 = (const unsigned*)(whn + (size_t)n0 * H_);
    unsigned* d0 = (unsigned*)l_whr;
    unsigned* d1 = (unsigned*)l_whz;
    unsigned* d2 = (unsigned*)l_whn;
    const int nd = HS * H_ / 2;
    for (int i = threadIdx.x; i < nd; i += blockDim.x) {
      d0[i] = s0[i]; d1[i] = s1[i]; d2[i] = s2[i];
    }
  }
  __syncthreads();

  const int wave = threadIdx.x >> 5;
  const int lane = threadIdx.x & 31;
  const int m0   = (wave & 1) * 16;       // batch-tile origin
  const int nn0  = (wave >> 1) * 16;      // channel-tile origin within slice
  const int ncol = lane & 15;
  const int moff = (lane >> 4) << 3;
  const int ng   = n0 + nn0 + ncol;       // this lane's global channel
  const float br = bhr[ng], bz = bhz[ng], bn = bhn[ng];

  unsigned bcount = 0;
  for (int t = 0; t < T_; ++t) {
    // ---- phase A: r and z (A operand = h, stable during this phase) ----
    v8f ar = z8(), az = z8();
    for (int k0 = 0; k0 < H_; k0 += 32) {
      v16bf a  = load_frag_a(hbf, H_, m0, k0, lane);
      v16bf b1 = load_frag_b(l_whr, H_, nn0, k0, lane);
      v16bf b2 = load_frag_b(l_whz, H_, nn0, k0, lane);
      ar = WMMA_BF16(a, b1, ar);
      az = WMMA_BF16(a, b2, az);
    }
    for (int rr = 0; rr < 8; ++rr) {
      int m = m0 + moff + rr;
      float pre = ar[rr] + xr[((size_t)m * T_ + t) * H_ + ng] + br;
      float r   = 1.0f / (1.0f + __expf(-pre));
      float rh  = r * hf[m * H_ + ng];
      rhbf[m * H_ + ng] = f2bf(rh);
    }
    gbar(bar, ++bcount * NWG);

    // ---- phase B: n gate (A operand = rh), state update ----
    v8f an = z8();
    for (int k0 = 0; k0 < H_; k0 += 32) {
      v16bf a = load_frag_a(rhbf, H_, m0, k0, lane);
      v16bf b = load_frag_b(l_whn, H_, nn0, k0, lane);
      an = WMMA_BF16(a, b, an);
    }
    for (int rr = 0; rr < 8; ++rr) {
      int m = m0 + moff + rr;
      float z    = 1.0f / (1.0f + __expf(-(az[rr] + xz[((size_t)m * T_ + t) * H_ + ng] + bz)));
      float nv   = tanhf(an[rr] + xn[((size_t)m * T_ + t) * H_ + ng] + bn);
      float hold = hf[m * H_ + ng];
      float hnew = (1.0f - z) * hold + z * nv;
      hf[m * H_ + ng]  = hnew;
      hbf[m * H_ + ng] = f2bf(hnew);
      out[((size_t)m * T_ + t) * H_ + ng] = hnew;
      if (t == T_ - 1) hn[m * H_ + ng] = hnew;
    }
    gbar(bar, ++bcount * NWG);
  }
}

// ------------------------------------------------------------------ launcher
extern "C" void kernel_launch(void* const* d_in, const int* in_sizes, int n_in,
                              void* d_out, int out_size, void* d_ws, size_t ws_size,
                              hipStream_t stream) {
  (void)in_sizes; (void)n_in; (void)out_size; (void)ws_size;
  const float* x   = (const float*)d_in[0];
  const float* Wir = (const float*)d_in[1];
  const float* Whr = (const float*)d_in[2];
  const float* bhr = (const float*)d_in[3];
  const float* Wiz = (const float*)d_in[4];
  const float* Whz = (const float*)d_in[5];
  const float* bhz = (const float*)d_in[6];
  const float* Win = (const float*)d_in[7];
  const float* Whn = (const float*)d_in[8];
  const float* bhn = (const float*)d_in[9];

  float* out = (float*)d_out;                    // [B,T,H]
  float* hn  = out + (size_t)B_ * T_ * H_;       // [B,H]

  char* ws = (char*)d_ws;
  size_t off = 0;
  auto alloc = [&](size_t bytes) -> void* {
    void* p = ws + off;
    off = (off + bytes + 255) & ~(size_t)255;
    return p;
  };
  unsigned short* xbf    = (unsigned short*)alloc((size_t)B_ * T_ * I_ * 2);
  unsigned short* wirb   = (unsigned short*)alloc((size_t)H_ * I_ * 2);
  unsigned short* wizb   = (unsigned short*)alloc((size_t)H_ * I_ * 2);
  unsigned short* winb   = (unsigned short*)alloc((size_t)H_ * I_ * 2);
  unsigned short* whrb   = (unsigned short*)alloc((size_t)H_ * H_ * 2);
  unsigned short* whzb   = (unsigned short*)alloc((size_t)H_ * H_ * 2);
  unsigned short* whnb   = (unsigned short*)alloc((size_t)H_ * H_ * 2);
  float* xrw = (float*)alloc((size_t)B_ * T_ * H_ * 4);
  float* xzw = (float*)alloc((size_t)B_ * T_ * H_ * 4);
  float* xnw = (float*)alloc((size_t)B_ * T_ * H_ * 4);
  char* state0 = ws + off;                       // zeroed region begins here
  float*          hf   = (float*)alloc((size_t)B_ * H_ * 4);
  unsigned short* hbf  = (unsigned short*)alloc((size_t)B_ * H_ * 2);
  unsigned short* rhbf = (unsigned short*)alloc((size_t)B_ * H_ * 2);
  unsigned*       bar  = (unsigned*)alloc(256);
  size_t state_bytes = (size_t)((ws + off) - state0);

  // f32 -> bf16 conversions
  const int nx = B_ * T_ * I_;
  const int nw = H_ * I_;
  cvt_f32_bf16<<<4096, 256, 0, stream>>>(x,   xbf,  nx);
  cvt_f32_bf16<<<1024, 256, 0, stream>>>(Wir, wirb, nw);
  cvt_f32_bf16<<<1024, 256, 0, stream>>>(Wiz, wizb, nw);
  cvt_f32_bf16<<<1024, 256, 0, stream>>>(Win, winb, nw);
  cvt_f32_bf16<<<1024, 256, 0, stream>>>(Whr, whrb, nw);
  cvt_f32_bf16<<<1024, 256, 0, stream>>>(Whz, whzb, nw);
  cvt_f32_bf16<<<1024, 256, 0, stream>>>(Whn, whnb, nw);

  // input projections: xr/xz/xn = x @ W^T
  gemm_inproj<<<dim3(2048, 3), 256, 0, stream>>>(xbf, wirb, wizb, winb, xrw, xzw, xnw);

  // zero h0 / rh / barrier counter (capture-safe)
  hipMemsetAsync(state0, 0, state_bytes, stream);

  // persistent recurrent scan (192 KB dynamic LDS per WG)
  gru_recurrent<<<NWG, 128, 3 * HS * H_ * 2, stream>>>(
      xrw, xzw, xnw, bhr, bhz, bhn, whrb, whzb, whnb,
      hf, hbf, rhbf, bar, out, hn);
}
